// Head_33457795236107
// MI455X (gfx1250) — matrix operand order repaired and checked by reading, not compiled, not run on hardware
//
#include <hip/hip_runtime.h>

// ---------------------------------------------------------------------------
// Fused head: q/k/v = x@W^T (+RoPE), flash attention with causal softmax.
// All GEMMs via v_wmma_f32_16x16x32_f16 (wave32, gfx1250).
// K/V^T staged to LDS via Tensor Data Mover (tensor_load_to_lds + TENSORcnt).
// ---------------------------------------------------------------------------

typedef _Float16 h16;
typedef __attribute__((ext_vector_type(16))) _Float16 v16h;
typedef __attribute__((ext_vector_type(8)))  _Float16 v8h;
typedef __attribute__((ext_vector_type(2)))  _Float16 v2h;
typedef __attribute__((ext_vector_type(8)))  float    v8f;
typedef __attribute__((ext_vector_type(4)))  float    v4f;
typedef __attribute__((ext_vector_type(4)))  unsigned int v4u;
typedef __attribute__((ext_vector_type(8)))  int      v8i;
typedef __attribute__((ext_vector_type(4)))  int      v4i;

namespace {
constexpr int kB  = 512;
constexpr int kT  = 256;
constexpr int kC  = 384;
constexpr int kHS = 64;

// workspace layout (units: halves)
constexpr size_t OFF_WQ = 0;
constexpr size_t OFF_WK = OFF_WQ + (size_t)kHS * kC;
constexpr size_t OFF_WV = OFF_WK + (size_t)kHS * kC;
constexpr size_t OFF_Q  = OFF_WV + (size_t)kHS * kC;          // (B,T,HS) f16
constexpr size_t OFF_K  = OFF_Q + (size_t)kB * kT * kHS;       // (B,T,HS) f16
constexpr size_t OFF_V  = OFF_K + (size_t)kB * kT * kHS;       // (B,HS,T) f16 (transposed)
// total ws: ~48.2 MiB

// LDS layout for attention kernel (units: halves); padded strides vs 64 banks
constexpr int K_STRIDE = 72;    // 64 + 8  (144 B rows = 128 B data + 16 B pad)
constexpr int V_STRIDE = 264;   // 256 + 8 (528 B rows = 512 B data + 16 B pad)
constexpr int P_STRIDE = 40;    // 32 + 8
constexpr int LDS_K = 0;
constexpr int LDS_V = kT * K_STRIDE;                 // 18432
constexpr int LDS_P = LDS_V + kHS * V_STRIDE;        // 35328
constexpr int LDS_HALVES = LDS_P + 8 * 16 * P_STRIDE; // 40448 halves = 80896 B
}  // namespace

static __device__ __forceinline__ v8f wmma_f16(v16h a, v16h b, v8f c) {
  // D = A(16x32) * B(32x16) + C, f32 accumulate
  return __builtin_amdgcn_wmma_f32_16x16x32_f16(false, a, false, b, (short)0, c,
                                                false, false);
}
static __device__ __forceinline__ v16h cat8(v8h lo, v8h hi) {
  return __builtin_shufflevector(lo, hi, 0, 1, 2, 3, 4, 5, 6, 7, 8, 9, 10, 11,
                                 12, 13, 14, 15);
}

#if __has_builtin(__builtin_amdgcn_tensor_load_to_lds)
#define USE_TDM 1
// Tensor DMA descriptor load of a 2D tile (whole tensor), element = 2 bytes,
// with LDS row padding (pad_interval/pad_amount per CDNA5 D# encoding).
// clang-23 signature: (uint32x4 g0, int32x8 g1, int32x4, int32x4, int32x8, i32 cpol)
static __device__ __forceinline__ void tdm_load_2d(unsigned lds_byte_off,
                                                   const void* gptr,
                                                   unsigned dim0, unsigned dim1,
                                                   unsigned stride0,
                                                   unsigned pad_interval,
                                                   unsigned pad_amount) {
  const unsigned long long ga = (unsigned long long)(uintptr_t)gptr;
  v4u g0;
  g0.x = 1u;                                            // count=1 (valid), user D#
  g0.y = lds_byte_off;                                  // lds_addr
  g0.z = (unsigned)ga;                                  // global_addr[31:0]
  g0.w = (unsigned)((ga >> 32) & 0x01FFFFFFu) | 0x80000000u;  // addr[56:32]|type=2
  v8i g1;
  g1[0] = (int)((1u << 16)                               // data_size = 2 bytes
                | (1u << 20)                             // pad_enable
                | (pad_interval << 22) | (pad_amount << 25));
  g1[1] = (int)((dim0 & 0xFFFFu) << 16);                 // abar_addr=0 | dim0 lo
  g1[2] = (int)(((dim0 >> 16) & 0xFFFFu) | ((dim1 & 0xFFFFu) << 16));
  g1[3] = (int)(((dim1 >> 16) & 0xFFFFu) | ((dim0 & 0xFFFFu) << 16));  // tile_dim0
  g1[4] = (int)(dim1 & 0xFFFFu);                         // tile_dim1 | tile_dim2=0
  g1[5] = (int)stride0;                                  // tensor_dim0_stride lo
  g1[6] = 0;                                             // stride0 hi | stride1 lo
  g1[7] = 0;
  const v4i z4 = {0, 0, 0, 0};                           // groups 2/3 unused (2D)
  const v8i z8 = {0, 0, 0, 0, 0, 0, 0, 0};
  __builtin_amdgcn_tensor_load_to_lds(g0, g1, z4, z4, z8, 0);
}
#else
#define USE_TDM 0
#endif

// ---------------------------------------------------------------------------
// Kernel 0: weights fp32 -> f16 (once; 3 x 64x384)
// ---------------------------------------------------------------------------
__global__ void k_cvt_w(const float* __restrict__ Wq, const float* __restrict__ Wk,
                        const float* __restrict__ Wv, h16* __restrict__ ws) {
  const int i = blockIdx.x * 256 + threadIdx.x;
  if (i < kHS * kC) {
    ws[OFF_WQ + i] = (h16)Wq[i];
    ws[OFF_WK + i] = (h16)Wk[i];
    ws[OFF_WV + i] = (h16)Wv[i];
  }
}

// ---------------------------------------------------------------------------
// Kernel 1: QKV projection + RoPE. One wave per 16-row tile of x.
//   A = x tile (16x32 per k-step, fp32->f16 in regs)
//   B = W rows, loaded directly in WMMA-B layout (B^T row-major == W row-major)
// ---------------------------------------------------------------------------
__global__ __launch_bounds__(256) void k_qkv_rope(
    const float* __restrict__ x, const h16* __restrict__ ws,
    h16* __restrict__ Qh, h16* __restrict__ Kh, h16* __restrict__ Vth) {
  const int lane  = threadIdx.x & 31;
  const int wid   = threadIdx.x >> 5;
  const int tile  = blockIdx.x * 8 + wid;   // 8192 tiles
  const int b     = tile >> 4;
  const int tbase = (tile & 15) << 4;
  const int lrow  = lane & 15;
  const int lhi   = lane >> 4;
  const int off8  = lhi * 8;

  const float* xrow = x + (size_t)(b * kT + tbase + lrow) * kC;

  v8f acc[3][4];  // [q,k,v][h-tile]
#pragma unroll
  for (int m = 0; m < 3; ++m)
#pragma unroll
    for (int n = 0; n < 4; ++n) acc[m][n] = (v8f){};

#pragma unroll
  for (int kc = 0; kc < 12; ++kc) {
    // A fragment: lane<16 holds K = {0..7,16..23}, lane>=16 holds {8..15,24..31}
    const int c0 = kc * 32 + off8;
    v4f x0 = *(const v4f*)(xrow + c0);
    v4f x1 = *(const v4f*)(xrow + c0 + 4);
    v4f x2 = *(const v4f*)(xrow + c0 + 16);
    v4f x3 = *(const v4f*)(xrow + c0 + 20);
    v16h a;
#pragma unroll
    for (int j = 0; j < 4; ++j) {
      a[j]      = (h16)x0[j];
      a[4 + j]  = (h16)x1[j];
      a[8 + j]  = (h16)x2[j];
      a[12 + j] = (h16)x3[j];
    }
#pragma unroll
    for (int n = 0; n < 4; ++n) {
      const size_t wrow = (size_t)(n * 16 + lrow) * kC + kc * 32 + off8;
#pragma unroll
      for (int m = 0; m < 3; ++m) {
        v16h bw = cat8(*(const v8h*)(ws + (size_t)m * kHS * kC + wrow), *(const v8h*)(ws + (size_t)m * kHS * kC + wrow + 16));
        acc[m][n] = wmma_f16(a, bw, acc[m][n]);
      }
    }
  }

  // RoPE (matches reference exactly):
  //   out[h]   = -q[h+1] * f(t, 32 + h/2)   (h even)
  //   out[h+1] =  q[h]   * f(t, h/2)
  //   f(t,c) = (c even ? sin : cos)(t * 10000^{-(c>>1)/32})
  // Even lane of each (h, h+1) pair computes BOTH outputs (it already has the
  // partner value from shfl_xor 1) and emits one packed 4-byte store.
  const float kLn1e4_32 = 0.28782313662425572f;  // ln(10000)/32
#pragma unroll
  for (int n = 0; n < 4; ++n) {
    const int h = n * 16 + lrow;                // C-layout: lane holds column h
    const int i = (h & ~1) >> 1;                // pair index (same for both lanes)
    const bool use_sin = ((i & 1) == 0);        // parity of c_e and c_o match
    const float dv_hi = __expf(-kLn1e4_32 * (float)(16 + (i >> 1)));  // c=32+i
    const float dv_lo = __expf(-kLn1e4_32 * (float)(i >> 1));         // c=i
#pragma unroll
    for (int r = 0; r < 8; ++r) {
      const int t = tbase + off8 + r;           // C-layout row
      const float ahi = (float)t * dv_hi;
      const float alo = (float)t * dv_lo;
      const float f_e = use_sin ? __sinf(ahi) : __cosf(ahi);
      const float f_o = use_sin ? __sinf(alo) : __cosf(alo);
      const float my_q = acc[0][n][r], my_k = acc[1][n][r];
      const float pw_q = __shfl_xor(my_q, 1);   // partner column h^1
      const float pw_k = __shfl_xor(my_k, 1);
      if (!(h & 1)) {                            // even lane stores the pair
        v2h pq, pk;
        pq[0] = (h16)(-pw_q * f_e);  pq[1] = (h16)(my_q * f_o);
        pk[0] = (h16)(-pw_k * f_e);  pk[1] = (h16)(my_k * f_o);
        const size_t o = (size_t)(b * kT + t) * kHS + h;   // h even -> 4B aligned
        *(v2h*)(Qh + o) = pq;
        *(v2h*)(Kh + o) = pk;
      }
    }
    // V stored transposed (h, t): 8 consecutive t per lane -> one 16B store
    v8h vv;
#pragma unroll
    for (int r = 0; r < 8; ++r) vv[r] = (h16)acc[2][n][r];
    *(v8h*)(Vth + (size_t)(b * kHS + h) * kT + tbase + off8) = vv;
  }
}

// ---------------------------------------------------------------------------
// Kernel 2: flash attention, one block (8 waves) per batch element.
//   S^T = K * Q^T  -> per-lane query column => per-lane softmax (+ shfl_xor 16)
//   O^T = V^T * P^T, P^T restaged via per-wave LDS tile in B layout
//   K/V^T staged by the Tensor Data Mover with hardware-inserted row padding.
// ---------------------------------------------------------------------------
__global__ __launch_bounds__(256) void k_attn(const h16* __restrict__ Qh,
                                              const h16* __restrict__ Kh,
                                              const h16* __restrict__ Vth,
                                              float* __restrict__ out) {
  extern __shared__ h16 lds[];
  const int b    = blockIdx.x;
  const int tid  = threadIdx.x;
  const int lane = tid & 31;
  const int wid  = tid >> 5;
  const int lrow = lane & 15;
  const int lhi  = lane >> 4;
  const int off8 = lhi * 8;
  const float NEG_INF = -__builtin_inff();

#if USE_TDM
  if (wid == 0) {
    // K: 256 rows x 64 halves; pad 4 DW after every 32 DW -> 144 B rows
    tdm_load_2d((unsigned)(uintptr_t)(lds + LDS_K),
                Kh + (size_t)b * kT * kHS, kHS, kT, kHS,
                /*pad_interval=*/4, /*pad_amount=*/3);
    // V^T: 64 rows x 256 halves; pad 4 DW after every 128 DW -> 528 B rows
    tdm_load_2d((unsigned)(uintptr_t)(lds + LDS_V),
                Vth + (size_t)b * kHS * kT, kT, kHS, kT,
                /*pad_interval=*/6, /*pad_amount=*/3);
    __builtin_amdgcn_s_wait_tensorcnt(0);
  }
#else
  {  // fallback: manual staged copy
    const v8h* srcK = (const v8h*)(Kh + (size_t)(b * kT + tid) * kHS);
    v8h* dstK = (v8h*)(lds + LDS_K + tid * K_STRIDE);
#pragma unroll
    for (int i = 0; i < 8; ++i) dstK[i] = srcK[i];
    const int vr = tid >> 2;
    const int vc = (tid & 3) * 64;
    const v8h* srcV = (const v8h*)(Vth + (size_t)(b * kHS + vr) * kT + vc);
    v8h* dstV = (v8h*)(lds + LDS_V + vr * V_STRIDE + vc);
#pragma unroll
    for (int i = 0; i < 8; ++i) dstV[i] = srcV[i];
  }
#endif
  __syncthreads();

  // Prefetch this wave's second q-tile rows while working on the first.
  __builtin_prefetch(Qh + (size_t)(b * kT + (wid + 8) * 16 + lrow) * kHS, 0, 1);

  h16* Pl = lds + LDS_P + wid * (16 * P_STRIDE);  // per-wave P^T scratch [q][s]

  for (int qi = 0; qi < 2; ++qi) {
    const int qt = wid + qi * 8;        // 16 q-tiles / 8 waves
    const int qbase = qt * 16;
    const int qg = qbase + lrow;        // this lane's query row (both halves)

    // Q fragments (B operand): lane reads Q row qbase+lrow, contiguous h
    const h16* qp = Qh + (size_t)(b * kT + qbase + lrow) * kHS + lhi * 16;
    const v16h bq0 = *(const v16h*)(qp);
    const v16h bq1 = *(const v16h*)(qp + 32);

    float m = NEG_INF, l = 0.f;
    v8f oacc[4];
#pragma unroll
    for (int n = 0; n < 4; ++n) oacc[n] = (v8f){};

    const int nb = (qbase + 47) >> 5;   // causal: ceil((qbase+16)/32) key blocks
    for (int kb = 0; kb < nb; ++kb) {
      const int kbase = kb * 32;
      // S^T tiles (16k x 16q), K-dim = HS = 2 x 32
      v8f s0 = (v8f){}, s1 = (v8f){};
#pragma unroll
      for (int hc = 0; hc < 2; ++hc) {
        const h16* k0 = lds + LDS_K + (kbase + lrow) * K_STRIDE + hc * 32 + lhi * 16;
        const h16* k1 = k0 + 16 * K_STRIDE;
        v16h aK0 = cat8(*(const v8h*)k0, *(const v8h*)(k0 + 8));
        v16h aK1 = cat8(*(const v8h*)k1, *(const v8h*)(k1 + 8));
        const v16h bq = hc ? bq1 : bq0;
        s0 = wmma_f16(aK0, bq, s0);
        s1 = wmma_f16(aK1, bq, s1);
      }
      // scale + causal mask + block max (per-lane, then pair-lane combine)
      const bool domask = (kbase + 31) > qbase;
      float bm = NEG_INF;
#pragma unroll
      for (int r = 0; r < 8; ++r) {
        float v0 = s0[r] * 0.125f;      // HS^-0.5
        float v1 = s1[r] * 0.125f;
        if (domask) {
          const int kg = kbase + off8 + r;
          if (kg > qg) v0 = NEG_INF;
          if (kg + 16 > qg) v1 = NEG_INF;
        }
        s0[r] = v0; s1[r] = v1;
        bm = fmaxf(bm, fmaxf(v0, v1));
      }
      bm = fmaxf(bm, __shfl_xor(bm, 16));
      const float mn  = fmaxf(m, bm);
      const float scl = __expf(m - mn);  // exp(-inf)=0 on first block
      float ps = 0.f;
      v8h p0, p1;
#pragma unroll
      for (int r = 0; r < 8; ++r) {
        const float e0 = __expf(s0[r] - mn);
        const float e1 = __expf(s1[r] - mn);
        ps += e0 + e1;
        p0[r] = (h16)e0;
        p1[r] = (h16)e1;
      }
      ps += __shfl_xor(ps, 16);
      l = l * scl + ps;
      m = mn;
#pragma unroll
      for (int n = 0; n < 4; ++n) oacc[n] *= scl;

      // P^T -> LDS [q][s] (same-wave DS ops are ordered; no barrier needed)
      *(v8h*)(Pl + lrow * P_STRIDE + off8) = p0;
      *(v8h*)(Pl + lrow * P_STRIDE + 16 + off8) = p1;
      const h16* pr = Pl + lrow * P_STRIDE + off8;
      const v16h bp = cat8(*(const v8h*)pr, *(const v8h*)(pr + 16));

      // O^T += V^T(16h x 32s) * P^T(32s x 16q), 4 h-tiles
#pragma unroll
      for (int n = 0; n < 4; ++n) {
        const h16* vp = lds + LDS_V + (n * 16 + lrow) * V_STRIDE + kbase + off8;
        v16h aV = cat8(*(const v8h*)vp, *(const v8h*)(vp + 16));
        oacc[n] = wmma_f16(aV, bp, oacc[n]);
      }
    }

    // epilogue: O[q][h] = oacc/l ; lane holds column q, 8 consecutive h per tile
    const float inv = 1.f / l;
    float* orow = out + (size_t)(b * kT + qbase + lrow) * kHS;
#pragma unroll
    for (int n = 0; n < 4; ++n) {
      v4f o0, o1;
#pragma unroll
      for (int j = 0; j < 4; ++j) {
        o0[j] = oacc[n][j] * inv;
        o1[j] = oacc[n][4 + j] * inv;
      }
      *(v4f*)(orow + n * 16 + off8) = o0;
      *(v4f*)(orow + n * 16 + off8 + 4) = o1;
    }
  }
}

// ---------------------------------------------------------------------------
extern "C" void kernel_launch(void* const* d_in, const int* in_sizes, int n_in,
                              void* d_out, int out_size, void* d_ws, size_t ws_size,
                              hipStream_t stream) {
  (void)in_sizes; (void)n_in; (void)out_size; (void)ws_size;
  const float* x  = (const float*)d_in[0];
  const float* Wq = (const float*)d_in[1];
  const float* Wk = (const float*)d_in[2];
  const float* Wv = (const float*)d_in[3];
  h16* ws  = (h16*)d_ws;
  h16* Qh  = ws + OFF_Q;
  h16* Kh  = ws + OFF_K;
  h16* Vth = ws + OFF_V;

  k_cvt_w<<<dim3(96), dim3(256), 0, stream>>>(Wq, Wk, Wv, ws);
  k_qkv_rope<<<dim3((kB * kT / 16) / 8), dim3(256), 0, stream>>>(x, ws, Qh, Kh, Vth);
  k_attn<<<dim3(kB), dim3(256), (unsigned)(LDS_HALVES * sizeof(h16)), stream>>>(
      Qh, Kh, Vth, (float*)d_out);
}
